// QwenAttention_23175643529290
// MI455X (gfx1250) — matrix operand order repaired and checked by reading, not compiled, not run on hardware
//
#include <hip/hip_runtime.h>

// ---------------------------------------------------------------------------
// CDNA5 (gfx1250) WMMA bf16 attention pipeline.
// All matrix math uses V_WMMA_F32_16X16X32_BF16 (wave32).
// Global->LDS staging uses GLOBAL_LOAD_ASYNC_TO_LDS_B128 when the toolchain
// exposes the builtin (guarded; falls back to load+ds_store otherwise).
// ---------------------------------------------------------------------------

typedef __attribute__((ext_vector_type(16))) __bf16 v16bf;
typedef __attribute__((ext_vector_type(8)))  float  v8f;
typedef __attribute__((ext_vector_type(4)))  unsigned int u32x4;
typedef __attribute__((ext_vector_type(4)))  int v4i;

#if defined(__has_builtin)
#if __has_builtin(__builtin_amdgcn_global_load_async_to_lds_b128) && \
    __has_builtin(__builtin_amdgcn_s_wait_asynccnt)
#define HAVE_ASYNC_LDS 1
#endif
#endif

#ifdef HAVE_ASYNC_LDS
typedef __attribute__((address_space(1))) v4i GlobalV4;   // prints as "__device__"
typedef __attribute__((address_space(3))) v4i LdsV4;      // prints as "__shared__"
// Per-lane 16-byte async copy: global -> LDS, tracked by ASYNCcnt.
__device__ __forceinline__ void async_cp16(const void* gptr, void* lptr) {
  __builtin_amdgcn_global_load_async_to_lds_b128((GlobalV4*)gptr, (LdsV4*)lptr,
                                                 0, 0);
}
__device__ __forceinline__ void async_wait0() {
  __builtin_amdgcn_s_wait_asynccnt(0);
}
#else
__device__ __forceinline__ void async_cp16(const void* gptr, void* lptr) {
  *(u32x4*)lptr = *(const u32x4*)gptr;
}
__device__ __forceinline__ void async_wait0() {}
#endif

union FragU { v16bf v; u32x4 q[2]; };

__device__ __forceinline__ unsigned short f2bf(float f) {
  unsigned int u = __float_as_uint(f);
  u += 0x7FFFu + ((u >> 16) & 1u);          // round-to-nearest-even
  return (unsigned short)(u >> 16);
}

// A-matrix fragment (16x32 bf16): lane<16 holds K{0..7,16..23}, lane>=16 K{8..15,24..31}
__device__ __forceinline__ v16bf ld_fragA(const unsigned short* base, int stride,
                                          int row0, int k0, int lane) {
  int r  = row0 + (lane & 15);
  int kb = k0 + ((lane & 16) ? 8 : 0);
  FragU t;
  t.q[0] = *(const u32x4*)(base + r * stride + kb);
  t.q[1] = *(const u32x4*)(base + r * stride + kb + 16);
  return t.v;
}

// B-matrix fragment from B^T rows (col n = lane&15): lane<16 K{0..7,16..23}, lane>=16 K{8..15,24..31}
__device__ __forceinline__ v16bf ld_fragB(const unsigned short* base, int stride,
                                          int row0, int k0, int lane) {
  int r  = row0 + (lane & 15);
  int kb = k0 + ((lane & 16) ? 8 : 0);
  FragU t;
  t.q[0] = *(const u32x4*)(base + r * stride + kb);
  t.q[1] = *(const u32x4*)(base + r * stride + kb + 16);
  return t.v;
}

__device__ __forceinline__ v8f wmma_bf16(v16bf a, v16bf b, v8f c) {
  return __builtin_amdgcn_wmma_f32_16x16x32_bf16(false, a, false, b,
                                                 (short)0, c, false, false);
}

// ---------------------------------------------------------------------------
// Elementwise fp32 -> bf16 conversion
// ---------------------------------------------------------------------------
__global__ __launch_bounds__(256) void cvt_bf16_kernel(const float* __restrict__ in,
                                                       unsigned short* __restrict__ out,
                                                       int n) {
  int i = blockIdx.x * blockDim.x + threadIdx.x;
  int stride = gridDim.x * blockDim.x;
  for (; i < n; i += stride) out[i] = f2bf(in[i]);
}

// ---------------------------------------------------------------------------
// Transpose + convert: out[n*R + r] = bf16(in[r*Ncols + n]); R, Ncols % 32 == 0
// ---------------------------------------------------------------------------
__global__ __launch_bounds__(256) void transpose_cvt_kernel(const float* __restrict__ in,
                                                            unsigned short* __restrict__ out,
                                                            int R, int Ncols) {
  __shared__ float tile[32][33];
  int n0 = blockIdx.x * 32, r0 = blockIdx.y * 32;
  #pragma unroll
  for (int j = 0; j < 4; ++j) {
    int r = r0 + threadIdx.y + j * 8;
    tile[threadIdx.y + j * 8][threadIdx.x] = in[(size_t)r * Ncols + n0 + threadIdx.x];
  }
  __syncthreads();
  #pragma unroll
  for (int j = 0; j < 4; ++j) {
    int n = n0 + threadIdx.y + j * 8;
    out[(size_t)n * R + r0 + threadIdx.x] = f2bf(tile[threadIdx.x][threadIdx.y + j * 8]);
  }
}

// ---------------------------------------------------------------------------
// Tiled bf16 WMMA GEMM: C[M,N] = A[M,K] * B[K,N] (+bias), B given transposed (Bt[N,K])
// BM=BN=128, BK=32; 8 waves in 4x2 grid, each wave owns a 32x64 tile (8 wmma accs).
// mode 0: write f32 row-major to outF
// mode 1: add bias, apply RoPE (head_dim==BN==128), write bf16 [b,h,t,d] to outB
// mode 2: add bias, write bf16 [b,h,t,d] to outB (no RoPE)
// ---------------------------------------------------------------------------
__global__ __launch_bounds__(256) void gemm_bf16_kernel(
    const unsigned short* __restrict__ A, int lda,
    const unsigned short* __restrict__ Bt, int ldb,
    const float* __restrict__ bias,
    float* __restrict__ outF, unsigned short* __restrict__ outB,
    int M, int N, int K, int mode,
    const float* __restrict__ cosT, const float* __restrict__ sinT,
    int Tlen, int Hh) {
  __shared__ unsigned short sA[128 * 40];
  __shared__ unsigned short sB[128 * 40];
  __shared__ float sC[128 * 132];

  const int tid  = threadIdx.x;
  const int lane = tid & 31;
  const int wave = tid >> 5;
  const int wm   = wave & 3;     // 0..3 -> 32-row slab
  const int wn   = wave >> 2;    // 0..1 -> 64-col slab
  const int m0   = blockIdx.y * 128;
  const int n0   = blockIdx.x * 128;

  v8f acc[2][4] = {};

  for (int k0 = 0; k0 < K; k0 += 32) {
    #pragma unroll
    for (int i = 0; i < 2; ++i) {
      int c  = tid + i * 256;          // 512 chunks of 8 bf16
      int r  = c >> 2;
      int cc = (c & 3) * 8;
      async_cp16(A + (size_t)(m0 + r) * lda + k0 + cc, &sA[r * 40 + cc]);
      async_cp16(Bt + (size_t)(n0 + r) * ldb + k0 + cc, &sB[r * 40 + cc]);
    }
    if (k0 + 32 < K) {   // global_prefetch_b8 of next K slab
      __builtin_prefetch(A + (size_t)(m0 + (tid >> 1)) * lda + k0 + 32, 0, 3);
      __builtin_prefetch(Bt + (size_t)(n0 + (tid >> 1)) * ldb + k0 + 32, 0, 3);
    }
    async_wait0();
    __syncthreads();
    v16bf af[2], bfv[4];
    #pragma unroll
    for (int mt = 0; mt < 2; ++mt) af[mt]  = ld_fragA(sA, 40, wm * 32 + mt * 16, 0, lane);
    #pragma unroll
    for (int nt = 0; nt < 4; ++nt) bfv[nt] = ld_fragB(sB, 40, wn * 64 + nt * 16, 0, lane);
    #pragma unroll
    for (int mt = 0; mt < 2; ++mt)
      #pragma unroll
      for (int nt = 0; nt < 4; ++nt)
        acc[mt][nt] = wmma_bf16(af[mt], bfv[nt], acc[mt][nt]);
    __syncthreads();
  }

  const int hi8 = (lane & 16) ? 8 : 0;
  if (mode == 0) {
    #pragma unroll
    for (int mt = 0; mt < 2; ++mt)
      #pragma unroll
      for (int nt = 0; nt < 4; ++nt)
        #pragma unroll
        for (int e = 0; e < 8; ++e) {
          int row = m0 + wm * 32 + mt * 16 + e + hi8;
          int col = n0 + wn * 64 + nt * 16 + (lane & 15);
          float v = acc[mt][nt][e];
          if (bias) v += bias[col];
          outF[(size_t)row * N + col] = v;
        }
  } else if (mode == 2) {
    #pragma unroll
    for (int mt = 0; mt < 2; ++mt)
      #pragma unroll
      for (int nt = 0; nt < 4; ++nt)
        #pragma unroll
        for (int e = 0; e < 8; ++e) {
          int rl  = wm * 32 + mt * 16 + e + hi8;
          int d   = wn * 64 + nt * 16 + (lane & 15);
          int row = m0 + rl;
          int b   = row / Tlen, t = row % Tlen;
          float v = acc[mt][nt][e];
          if (bias) v += bias[n0 + d];
          outB[(((size_t)(b * Hh + (int)blockIdx.x)) * Tlen + t) * 128 + d] = f2bf(v);
        }
  } else {  // mode 1: bias + RoPE, BN == head_dim == 128
    #pragma unroll
    for (int mt = 0; mt < 2; ++mt)
      #pragma unroll
      for (int nt = 0; nt < 4; ++nt)
        #pragma unroll
        for (int e = 0; e < 8; ++e) {
          int rl = wm * 32 + mt * 16 + e + hi8;
          int cl = wn * 64 + nt * 16 + (lane & 15);
          float v = acc[mt][nt][e];
          if (bias) v += bias[n0 + cl];
          sC[rl * 132 + cl] = v;
        }
    __syncthreads();
    for (int i = 0; i < 64; ++i) {
      int idx = tid + i * 256;
      int rl  = idx >> 7;
      int d   = idx & 127;
      int row = m0 + rl;
      int b   = row / Tlen, t = row % Tlen;
      float val = sC[rl * 132 + d];
      float rot = (d < 64) ? -sC[rl * 132 + d + 64] : sC[rl * 132 + d - 64];
      float cv  = cosT[(size_t)t * 128 + d];
      float sv  = sinT[(size_t)t * 128 + d];
      outB[(((size_t)(b * Hh + (int)blockIdx.x)) * Tlen + t) * 128 + d] =
          f2bf(val * cv + rot * sv);
    }
  }
}

// ---------------------------------------------------------------------------
// Flash attention (causal, GQA 4:1). Per block: 128 Q rows x one head.
// KV tiles of 64. S = Q*K^T and O += P*V via WMMA bf16; online softmax in f32.
// Q/K/V are bf16 in [b, h, t, 128] layout; output bf16 [b*T + t, h*128 + d].
// ---------------------------------------------------------------------------
__global__ __launch_bounds__(256) void attn_kernel(
    const unsigned short* __restrict__ Qb, const unsigned short* __restrict__ Kb,
    const unsigned short* __restrict__ Vb, unsigned short* __restrict__ Ob,
    int T, int H, int KVH) {
  __shared__ unsigned short sQ[128 * 136];   // Q tile  [row][d]
  __shared__ unsigned short sK[64 * 136];    // K tile  [kv][d]   (== B^T for S)
  __shared__ unsigned short sVt[128 * 72];   // V^T     [d][kv]   (== B^T for O)
  __shared__ float          sS[128 * 68];    // scores  [row][kv]
  __shared__ unsigned short sP[128 * 72];    // probs   [row][kv]
  __shared__ float c_sh[128];
  __shared__ float l_sh[128];

  const int qt = blockIdx.x, h = blockIdx.y, b = blockIdx.z;
  const int kh = h >> 2;                      // N_GROUPS = 4
  const int tid = threadIdx.x, lane = tid & 31, wave = tid >> 5;
  const int wm = wave & 3, wn = wave >> 2;
  const int hi8 = (lane & 16) ? 8 : 0;
  const int q0 = qt * 128;
  const float scale = 0.08838834764831845f;   // 1/sqrt(128)

  const unsigned short* Qp = Qb + (((size_t)(b * H + h)) * T + q0) * 128;
  const unsigned short* Kp = Kb + (((size_t)(b * KVH + kh)) * T) * 128;
  const unsigned short* Vp = Vb + (((size_t)(b * KVH + kh)) * T) * 128;

  #pragma unroll
  for (int i = 0; i < 8; ++i) {               // stage Q: 128x128 bf16 (async)
    int c = tid + i * 256;
    int r = c >> 4, cc = (c & 15) * 8;
    async_cp16(Qp + (size_t)r * 128 + cc, &sQ[r * 136 + cc]);
  }

  v8f o[2][4] = {};
  float m_run = -1e30f, l_run = 0.0f;

  const int njt = 2 * qt + 2;                 // causal: kv tiles with k0 < q0+128
  for (int jt = 0; jt < njt; ++jt) {
    const int k0 = jt * 64;
    __syncthreads();                          // previous sK/sVt/sP consumers done
    #pragma unroll
    for (int i = 0; i < 4; ++i) {             // stage K tile 64x128 (async)
      int c = tid + i * 256;
      int r = c >> 4, cc = (c & 15) * 8;
      async_cp16(Kp + (size_t)(k0 + r) * 128 + cc, &sK[r * 136 + cc]);
    }
    #pragma unroll
    for (int i = 0; i < 4; ++i) {             // stage V tile transposed -> sVt[d][kv]
      int c = tid + i * 256;
      int r = c >> 4, cc = (c & 15) * 8;
      u32x4 vv = *(const u32x4*)(Vp + (size_t)(k0 + r) * 128 + cc);
      const unsigned short* pv = (const unsigned short*)&vv;
      #pragma unroll
      for (int e = 0; e < 8; ++e) sVt[(cc + e) * 72 + r] = pv[e];
    }
    async_wait0();
    __syncthreads();

    // S = Q * K^T  (128 x 64), wave tile 32x32
    v8f sacc[2][2] = {};
    #pragma unroll
    for (int kk = 0; kk < 128; kk += 32) {
      v16bf aq[2], bk2[2];
      #pragma unroll
      for (int mt = 0; mt < 2; ++mt) aq[mt]  = ld_fragA(sQ, 136, wm * 32 + mt * 16, kk, lane);
      #pragma unroll
      for (int nt = 0; nt < 2; ++nt) bk2[nt] = ld_fragB(sK, 136, wn * 32 + nt * 16, kk, lane);
      #pragma unroll
      for (int mt = 0; mt < 2; ++mt)
        #pragma unroll
        for (int nt = 0; nt < 2; ++nt)
          sacc[mt][nt] = wmma_bf16(aq[mt], bk2[nt], sacc[mt][nt]);
    }
    #pragma unroll
    for (int mt = 0; mt < 2; ++mt)
      #pragma unroll
      for (int nt = 0; nt < 2; ++nt)
        #pragma unroll
        for (int e = 0; e < 8; ++e) {
          int rl = wm * 32 + mt * 16 + e + hi8;
          int cl = wn * 32 + nt * 16 + (lane & 15);
          sS[rl * 68 + cl] = sacc[mt][nt][e];
        }
    __syncthreads();

    // Online softmax: one row per thread (tid < 128)
    if (tid < 128) {
      int i = tid, qg = q0 + i;
      float tm = -1e30f;
      for (int j = 0; j < 64; ++j) {
        float s = sS[i * 68 + j] * scale;
        s = ((k0 + j) <= qg) ? s : -1e30f;
        tm = fmaxf(tm, s);
      }
      float mn   = fmaxf(m_run, tm);
      float corr = __expf(m_run - mn);
      float sum  = 0.0f;
      for (int j = 0; j < 64; ++j) {
        float s = sS[i * 68 + j] * scale;
        float p = ((k0 + j) <= qg) ? __expf(s - mn) : 0.0f;
        sum += p;
        sP[i * 72 + j] = f2bf(p);
      }
      m_run = mn;
      l_run = l_run * corr + sum;
      c_sh[i] = corr;
    }
    __syncthreads();

    // O = O*corr + P*V  (128 x 128 over kv=64)
    #pragma unroll
    for (int mt = 0; mt < 2; ++mt)
      #pragma unroll
      for (int e = 0; e < 8; ++e) {
        int rl = wm * 32 + mt * 16 + e + hi8;
        float cf = c_sh[rl];
        #pragma unroll
        for (int nt = 0; nt < 4; ++nt) o[mt][nt][e] *= cf;
      }
    #pragma unroll
    for (int kk = 0; kk < 64; kk += 32) {
      v16bf ap[2], bv4[4];
      #pragma unroll
      for (int mt = 0; mt < 2; ++mt) ap[mt]  = ld_fragA(sP, 72, wm * 32 + mt * 16, kk, lane);
      #pragma unroll
      for (int nt = 0; nt < 4; ++nt) bv4[nt] = ld_fragB(sVt, 72, wn * 64 + nt * 16, kk, lane);
      #pragma unroll
      for (int mt = 0; mt < 2; ++mt)
        #pragma unroll
        for (int nt = 0; nt < 4; ++nt)
          o[mt][nt] = wmma_bf16(ap[mt], bv4[nt], o[mt][nt]);
    }
  }

  __syncthreads();
  if (tid < 128) l_sh[tid] = l_run;
  __syncthreads();

  #pragma unroll
  for (int mt = 0; mt < 2; ++mt)
    #pragma unroll
    for (int nt = 0; nt < 4; ++nt)
      #pragma unroll
      for (int e = 0; e < 8; ++e) {
        int rl = wm * 32 + mt * 16 + e + hi8;
        int cl = wn * 64 + nt * 16 + (lane & 15);
        int t  = q0 + rl;
        float ov = o[mt][nt][e] / l_sh[rl];
        Ob[(size_t)(b * T + t) * (H * 128) + h * 128 + cl] = f2bf(ov);
      }
}

// ---------------------------------------------------------------------------
// Host launch
// ---------------------------------------------------------------------------
extern "C" void kernel_launch(void* const* d_in, const int* in_sizes, int n_in,
                              void* d_out, int out_size, void* d_ws, size_t ws_size,
                              hipStream_t stream) {
  (void)in_sizes; (void)n_in; (void)out_size; (void)ws_size;
  const float* hs   = (const float*)d_in[0];
  // d_in[1] = mask (all true; causal handled explicitly)
  const float* cosT = (const float*)d_in[2];
  const float* sinT = (const float*)d_in[3];
  const float* Wq   = (const float*)d_in[4];
  const float* bq   = (const float*)d_in[5];
  const float* Wk   = (const float*)d_in[6];
  const float* bk   = (const float*)d_in[7];
  const float* Wv   = (const float*)d_in[8];
  const float* bv   = (const float*)d_in[9];
  const float* Wo   = (const float*)d_in[10];

  const int Bn = 2, T = 2048, C = 2048, H = 16, KVH = 4, D = 128;
  const int M = Bn * T;                 // 4096
  const int Nq = H * D, Nkv = KVH * D;  // 2048, 512

  size_t off = 0;
  auto alloc = [&](size_t bytes) -> void* {
    void* p = (char*)d_ws + off;
    off += (bytes + 255) & ~(size_t)255;
    return p;
  };
  unsigned short* hsb = (unsigned short*)alloc((size_t)M * C * 2);
  unsigned short* wqT = (unsigned short*)alloc((size_t)Nq * C * 2);
  unsigned short* wkT = (unsigned short*)alloc((size_t)Nkv * C * 2);
  unsigned short* wvT = (unsigned short*)alloc((size_t)Nkv * C * 2);
  unsigned short* woT = (unsigned short*)alloc((size_t)C * Nq * 2);
  unsigned short* Qb  = (unsigned short*)alloc((size_t)M * Nq * 2);
  unsigned short* Kb  = (unsigned short*)alloc((size_t)M * Nkv * 2);
  unsigned short* Vb  = (unsigned short*)alloc((size_t)M * Nkv * 2);
  unsigned short* Ob  = (unsigned short*)alloc((size_t)M * Nq * 2);

  cvt_bf16_kernel<<<2048, 256, 0, stream>>>(hs, hsb, M * C);
  transpose_cvt_kernel<<<dim3(Nq / 32, C / 32), dim3(32, 8), 0, stream>>>(Wq, wqT, C, Nq);
  transpose_cvt_kernel<<<dim3(Nkv / 32, C / 32), dim3(32, 8), 0, stream>>>(Wk, wkT, C, Nkv);
  transpose_cvt_kernel<<<dim3(Nkv / 32, C / 32), dim3(32, 8), 0, stream>>>(Wv, wvT, C, Nkv);
  transpose_cvt_kernel<<<dim3(C / 32, Nq / 32), dim3(32, 8), 0, stream>>>(Wo, woT, Nq, C);

  // Q/K projections with fused bias + RoPE -> bf16 [b,h,t,d]
  gemm_bf16_kernel<<<dim3(Nq / 128, M / 128), 256, 0, stream>>>(
      hsb, C, wqT, C, bq, nullptr, Qb, M, Nq, C, 1, cosT, sinT, T, H);
  gemm_bf16_kernel<<<dim3(Nkv / 128, M / 128), 256, 0, stream>>>(
      hsb, C, wkT, C, bk, nullptr, Kb, M, Nkv, C, 1, cosT, sinT, T, KVH);
  // V projection -> bf16 [b,h,t,d]
  gemm_bf16_kernel<<<dim3(Nkv / 128, M / 128), 256, 0, stream>>>(
      hsb, C, wvT, C, bv, nullptr, Vb, M, Nkv, C, 2, nullptr, nullptr, T, KVH);

  attn_kernel<<<dim3(T / 128, H, Bn), 256, 0, stream>>>(Qb, Kb, Vb, Ob, T, H, KVH);

  // Output projection -> f32 d_out
  gemm_bf16_kernel<<<dim3(C / 128, M / 128), 256, 0, stream>>>(
      Ob, Nq, woT, Nq, nullptr, (float*)d_out, nullptr, M, C, Nq, 0,
      nullptr, nullptr, T, 0);
}
